// learnable_aa_24919400251506
// MI455X (gfx1250) — compile-verified, matrix operand order checked by reading, not run on hardware
//
#include <hip/hip_runtime.h>
#include <math.h>

#define D_DIM  2048
#define BSZ    2048
#define MHIST  6
#define THRESH 30
#define LAM    0.0001f
#define BETA   1.0f

typedef __attribute__((ext_vector_type(16))) __bf16 v16bf;
typedef __attribute__((ext_vector_type(8)))  float  v8f;

union FragBF { v16bf v; uint4 q[2]; unsigned int u[8]; };

__device__ __forceinline__ unsigned short f2bf(float f) {
    union { float f; unsigned int u; } cv; cv.f = f;
    unsigned int u = cv.u;
    unsigned int rounding = 0x7FFFu + ((u >> 16) & 1u);
    return (unsigned short)((u + rounding) >> 16);
}

// ---------------- small utility kernels ----------------

__global__ void zero_norms_kernel(float* norms) {
    if (threadIdx.x < 2 * THRESH) norms[threadIdx.x] = 0.0f;
}

__global__ void convert_w_kernel(const float* __restrict__ W,
                                 unsigned short* __restrict__ Wbf) {
    size_t idx = (size_t)blockIdx.x * blockDim.x + threadIdx.x;
    if (idx < (size_t)D_DIM * D_DIM) Wbf[idx] = f2bf(W[idx]);
}

// X[:,0] = x0 ; xbf = bf16(x0)
__global__ void init0_kernel(const float* __restrict__ x0,
                             float* __restrict__ Xr,
                             unsigned short* __restrict__ xbf) {
    size_t idx = (size_t)blockIdx.x * blockDim.x + threadIdx.x;
    if (idx >= (size_t)BSZ * D_DIM) return;
    size_t b  = idx >> 11;          // / D_DIM
    size_t dd = idx & (D_DIM - 1);
    float v = x0[idx];
    Xr[b * (size_t)(MHIST * D_DIM) + dd] = v;   // slot 0
    xbf[idx] = f2bf(v);
}

__global__ void copy_result_kernel(const float* __restrict__ Xr,
                                   float* __restrict__ out0, int lastSlot) {
    size_t idx = (size_t)blockIdx.x * blockDim.x + threadIdx.x;
    if (idx >= (size_t)BSZ * D_DIM) return;
    size_t b  = idx >> 11;
    size_t dd = idx & (D_DIM - 1);
    out0[idx] = Xr[(b * MHIST + lastSlot) * (size_t)D_DIM + dd];
}

__global__ void finalize_kernel(const float* __restrict__ norms,
                                float* __restrict__ rel,
                                float* __restrict__ absd) {
    int t = threadIdx.x;
    if (t < THRESH - 1) {
        int k = t + 1;
        float a = sqrtf(norms[2 * k]);
        float f = sqrtf(norms[2 * k + 1]);
        absd[t] = a;
        rel[t]  = a / (1e-5f + f);
    }
}

// ---------------- per-batch-row bordered solve ----------------
// One wave32 per batch row: compute GGt (pair dot products), build the
// bordered (n+1)x(n+1) system, pivoted Gaussian elimination on lane 0.
__global__ __launch_bounds__(256)
void solve_kernel(const float* __restrict__ Xr, const float* __restrict__ F,
                  float* __restrict__ alpha, int n) {
    const int lane = threadIdx.x & 31;
    const int row  = blockIdx.x * 8 + (threadIdx.x >> 5);  // batch index
    const float* xb = Xr + (size_t)row * (MHIST * D_DIM);
    const float* fb = F  + (size_t)row * (MHIST * D_DIM);

    float acc[21];
    #pragma unroll
    for (int p = 0; p < 21; ++p) acc[p] = 0.0f;

    for (int dd = lane; dd < D_DIM; dd += 32) {
        float g[MHIST];
        for (int i = 0; i < n; ++i)
            g[i] = fb[(size_t)i * D_DIM + dd] - xb[(size_t)i * D_DIM + dd];
        int p = 0;
        for (int i = 0; i < n; ++i)
            for (int j = 0; j <= i; ++j)
                acc[p++] += g[i] * g[j];
    }

    const int np = n * (n + 1) / 2;
    for (int p = 0; p < np; ++p)
        for (int o = 16; o; o >>= 1)
            acc[p] += __shfl_xor(acc[p], o, 32);

    if (lane == 0) {
        const int nn = n + 1;          // bordered system size (<= 7)
        float H[7][8];                 // augmented [7 x 8]
        for (int r = 0; r < nn; ++r)
            for (int c = 0; c <= nn; ++c) H[r][c] = 0.0f;
        for (int j = 1; j < nn; ++j) { H[0][j] = 1.0f; H[j][0] = 1.0f; }
        for (int i = 0; i < n; ++i)
            for (int j = 0; j <= i; ++j) {
                float v = acc[i * (i + 1) / 2 + j];
                H[1 + i][1 + j] = v;
                H[1 + j][1 + i] = v;
            }
        for (int i = 0; i < n; ++i) H[1 + i][1 + i] += LAM;
        H[0][nn] = 1.0f;               // rhs = e0

        // Gaussian elimination with partial pivoting
        for (int c = 0; c < nn; ++c) {
            int piv = c; float best = fabsf(H[c][c]);
            for (int r = c + 1; r < nn; ++r) {
                float m = fabsf(H[r][c]);
                if (m > best) { best = m; piv = r; }
            }
            if (piv != c)
                for (int j = c; j <= nn; ++j) {
                    float t = H[c][j]; H[c][j] = H[piv][j]; H[piv][j] = t;
                }
            float inv = 1.0f / H[c][c];
            for (int r = c + 1; r < nn; ++r) {
                float fct = H[r][c] * inv;
                for (int j = c; j <= nn; ++j) H[r][j] -= fct * H[c][j];
            }
        }
        float z[7];
        for (int r = nn - 1; r >= 0; --r) {
            float s = H[r][nn];
            for (int j = r + 1; j < nn; ++j) s -= H[r][j] * z[j];
            z[r] = s / H[r][r];
        }
        for (int i = 0; i < n; ++i) alpha[(size_t)row * 8 + i] = z[i + 1];
    }
}

// ---------------- xnew = beta*A@F + (1-beta)*A@X ; write X[:,up] + bf16 copy
__global__ __launch_bounds__(256)
void mix_kernel(float* __restrict__ Xr, const float* __restrict__ F,
                const float* __restrict__ alpha,
                unsigned short* __restrict__ xbf, int n, int up) {
    size_t idx = (size_t)blockIdx.x * blockDim.x + threadIdx.x;
    if (idx >= (size_t)BSZ * D_DIM) return;
    size_t b  = idx >> 11;
    size_t dd = idx & (D_DIM - 1);
    const float* al = alpha + b * 8;
    float sf = 0.0f, sx = 0.0f;
    for (int i = 0; i < n; ++i) {
        float a = al[i];
        size_t off = (b * MHIST + i) * (size_t)D_DIM + dd;
        sf += a * F[off];
        sx += a * Xr[off];
    }
    float xn = BETA * sf + (1.0f - BETA) * sx;
    Xr[(b * MHIST + up) * (size_t)D_DIM + dd] = xn;
    xbf[idx] = f2bf(xn);
}

// ---------------- bf16 WMMA GEMM + bias + tanh + norm reduction ----------------
// C(2048x2048) = tanh(A@W + b); block tile 128x128, 8 waves (2x4), wave tile
// 64x32 = 4x2 WMMA 16x16 tiles, BK = 32. Double-buffered LDS (1 barrier/step).
// A-tile staged with gfx1250 async copy (global_load_async_to_lds_b128,
// ASYNCcnt); B-tile staged with regular loads + transposed DS stores so the
// B fragments become two contiguous ds_load_b128 per lane.
__global__ __launch_bounds__(256)
void gemm_tanh_kernel(const unsigned short* __restrict__ Abf,  // [BSZ x D] bf16
                      const unsigned short* __restrict__ Wbf,  // [D x D] bf16
                      const float* __restrict__ bias,          // [D]
                      float* __restrict__ Fbase,   // F + up*D, row stride 6*D
                      const float* __restrict__ Xup, // X + up*D, row stride 6*D
                      float* __restrict__ normAcc,   // [2] accumulators
                      int doNorm) {
    __shared__ __align__(16) unsigned short lA[2][128 * 32];  // [m][k]
    __shared__ __align__(16) unsigned short lB[2][128 * 32];  // [n][k] transposed

    const int tid   = threadIdx.x;
    const int lane  = tid & 31;
    const int wid   = tid >> 5;
    const int waveM = wid & 1;        // 2 waves along M (64 rows each)
    const int waveN = wid >> 1;       // 4 waves along N (32 cols each)
    const int row0  = blockIdx.y * 128;
    const int col0  = blockIdx.x * 128;

    const int half = lane >> 4;       // lanes 16-31 hold the upper K pattern
    const int ml   = lane & 15;

    // cooperative tile-load coordinates
    const int am = tid >> 1;            // 0..127   (A row in tile)
    const int ak = (tid & 1) * 16;      // 0 or 16  (A k-segment)
    const int bk = tid >> 3;            // 0..31    (W row in tile)
    const int bn = (tid & 7) * 16;      // 0..112   (W n-segment)

    v8f c[4][2];
    #pragma unroll
    for (int tm = 0; tm < 4; ++tm)
        #pragma unroll
        for (int tn = 0; tn < 2; ++tn)
            c[tm][tn] = (v8f)(0.0f);

    // ---- prologue: stage tile k0=0 into buffer 0 ----
    {
        // A: async copy 32B per thread (two b128 beats, offset applies to
        // both LDS and global sides per ISA pseudocode)
        unsigned ldsA = (unsigned)(uintptr_t)&lA[0][am * 32 + ak];
        unsigned long long gA =
            (unsigned long long)(uintptr_t)(Abf + (size_t)(row0 + am) * D_DIM + ak);
        asm volatile("global_load_async_to_lds_b128 %0, %1, off"
                     :: "v"(ldsA), "v"(gA) : "memory");
        asm volatile("global_load_async_to_lds_b128 %0, %1, off offset:16"
                     :: "v"(ldsA), "v"(gA) : "memory");
        // B: load 16 bf16 along N, scatter transposed into [n][k]
        const uint4* gB = (const uint4*)(Wbf + (size_t)bk * D_DIM + col0 + bn);
        uint4 b0 = gB[0], b1 = gB[1];
        unsigned short bs[16];
        *(uint4*)&bs[0] = b0; *(uint4*)&bs[8] = b1;
        #pragma unroll
        for (int j = 0; j < 16; ++j) lB[0][(bn + j) * 32 + bk] = bs[j];
        asm volatile("s_wait_asynccnt 0x0" ::: "memory");
        __syncthreads();
    }

    int cur = 0;
    for (int k0 = 0; k0 < D_DIM; k0 += 32) {
        const int nxt = cur ^ 1;
        const bool hasNext = (k0 + 32 < D_DIM);
        uint4 b0, b1;
        if (hasNext) {
            // A-tile for next step: async global -> LDS (other buffer)
            unsigned ldsA = (unsigned)(uintptr_t)&lA[nxt][am * 32 + ak];
            unsigned long long gA = (unsigned long long)(uintptr_t)
                (Abf + (size_t)(row0 + am) * D_DIM + (k0 + 32) + ak);
            asm volatile("global_load_async_to_lds_b128 %0, %1, off"
                         :: "v"(ldsA), "v"(gA) : "memory");
            asm volatile("global_load_async_to_lds_b128 %0, %1, off offset:16"
                         :: "v"(ldsA), "v"(gA) : "memory");
            // B-tile for next step: global -> regs
            const uint4* gB =
                (const uint4*)(Wbf + (size_t)(k0 + 32 + bk) * D_DIM + col0 + bn);
            b0 = gB[0]; b1 = gB[1];
        }

        // ---- fragments from current buffer: 2x ds_load_b128 each ----
        const unsigned short* Ac = lA[cur];
        const unsigned short* Bc = lB[cur];
        FragBF afr[4], bfr[2];
        #pragma unroll
        for (int tm = 0; tm < 4; ++tm) {
            int rb = (waveM * 64 + tm * 16 + ml) * 32 + half * 8;
            afr[tm].q[0] = *(const uint4*)&Ac[rb];
            afr[tm].q[1] = *(const uint4*)&Ac[rb + 16];
        }
        #pragma unroll
        for (int tn = 0; tn < 2; ++tn) {
            int rb = (waveN * 32 + tn * 16 + ml) * 32 + half * 8;
            bfr[tn].q[0] = *(const uint4*)&Bc[rb];
            bfr[tn].q[1] = *(const uint4*)&Bc[rb + 16];
        }
        #pragma unroll
        for (int tm = 0; tm < 4; ++tm)
            #pragma unroll
            for (int tn = 0; tn < 2; ++tn)
                c[tm][tn] = __builtin_amdgcn_wmma_f32_16x16x32_bf16(
                    false, afr[tm].v, false, bfr[tn].v,
                    (short)0, c[tm][tn], false, false);

        if (hasNext) {
            // transpose-store B regs into the other buffer
            unsigned short bs[16];
            *(uint4*)&bs[0] = b0; *(uint4*)&bs[8] = b1;
            #pragma unroll
            for (int j = 0; j < 16; ++j) lB[nxt][(bn + j) * 32 + bk] = bs[j];
            asm volatile("s_wait_asynccnt 0x0" ::: "memory");
            __syncthreads();
            cur = nxt;
        }
    }

    // epilogue: bias + tanh, write F slot, accumulate norms
    const size_t rstride = (size_t)MHIST * D_DIM;
    float s_abs = 0.0f, s_fn = 0.0f;
    #pragma unroll
    for (int tm = 0; tm < 4; ++tm) {
        #pragma unroll
        for (int tn = 0; tn < 2; ++tn) {
            int col = col0 + waveN * 32 + tn * 16 + ml;
            float bia = bias[col];
            #pragma unroll
            for (int i = 0; i < 8; ++i) {
                int r = row0 + waveM * 64 + tm * 16 + i + 8 * half;
                float f = tanhf(c[tm][tn][i] + bia);
                size_t off = (size_t)r * rstride + col;
                Fbase[off] = f;
                if (doNorm) {
                    float d = f - Xup[off];
                    s_abs += d * d;
                    s_fn  += f * f;
                }
            }
        }
    }
    if (doNorm) {
        for (int o = 16; o; o >>= 1) {
            s_abs += __shfl_xor(s_abs, o, 32);
            s_fn  += __shfl_xor(s_fn, o, 32);
        }
        if (lane == 0) {
            atomicAdd(&normAcc[0], s_abs);
            atomicAdd(&normAcc[1], s_fn);
        }
    }
}

// ---------------- host driver ----------------

extern "C" void kernel_launch(void* const* d_in, const int* in_sizes, int n_in,
                              void* d_out, int out_size, void* d_ws, size_t ws_size,
                              hipStream_t stream) {
    const float* x0   = (const float*)d_in[0];
    const float* W    = (const float*)d_in[1];
    const float* bias = (const float*)d_in[2];

    // output layout: [result BSZ*D][X BSZ*6*D][rel 29][absd 29], all fp32
    float* out0 = (float*)d_out;
    float* Xr   = out0 + (size_t)BSZ * D_DIM;
    float* rel  = Xr + (size_t)BSZ * MHIST * D_DIM;
    float* absd = rel + (THRESH - 1);

    // workspace layout
    char* ws = (char*)d_ws;
    unsigned short* Wbf = (unsigned short*)ws;                       // 8 MB
    unsigned short* xbf = (unsigned short*)(ws + (size_t)8  * 1024 * 1024); // 8 MB
    float* F     = (float*)(ws + (size_t)16 * 1024 * 1024);          // 96 MB
    float* alpha = (float*)(ws + (size_t)16 * 1024 * 1024
                               + (size_t)BSZ * MHIST * D_DIM * 4);
    float* norms = alpha + (size_t)BSZ * 8;

    const int elemBlocks = (BSZ * D_DIM) / 256;       // 16384
    const dim3 gemmGrid(D_DIM / 128, BSZ / 128);      // 16 x 16

    zero_norms_kernel<<<1, 64, 0, stream>>>(norms);
    convert_w_kernel<<<(D_DIM * D_DIM) / 256, 256, 0, stream>>>(W, Wbf);
    init0_kernel<<<elemBlocks, 256, 0, stream>>>(x0, Xr, xbf);

    // F[:,0] = tanh(x0 @ W + b)
    gemm_tanh_kernel<<<gemmGrid, 256, 0, stream>>>(
        xbf, Wbf, bias, F, Xr, norms, 0);

    int up = 0;
    for (int k = 1; k < THRESH; ++k) {
        const int n = (k < MHIST) ? k : MHIST;
        up = k % MHIST;
        solve_kernel<<<BSZ / 8, 256, 0, stream>>>(Xr, F, alpha, n);
        mix_kernel<<<elemBlocks, 256, 0, stream>>>(Xr, F, alpha, xbf, n, up);
        gemm_tanh_kernel<<<gemmGrid, 256, 0, stream>>>(
            xbf, Wbf, bias,
            F + (size_t)up * D_DIM,
            Xr + (size_t)up * D_DIM,
            norms + 2 * k, 1);
    }

    finalize_kernel<<<1, 32, 0, stream>>>(norms, rel, absd);
    copy_result_kernel<<<elemBlocks, 256, 0, stream>>>(Xr, out0, up);
}
